// NNUENetwork_17420387352988
// MI455X (gfx1250) — matrix operand order repaired and checked by reading, not compiled, not run on hardware
//
#include <hip/hip_runtime.h>
#include <cstddef>

typedef __attribute__((ext_vector_type(16))) __bf16 v16bf;
typedef __attribute__((ext_vector_type(8)))  float  v8f;
typedef __attribute__((ext_vector_type(4)))  float  f32x4;
typedef __attribute__((ext_vector_type(4)))  unsigned int u32x4;

namespace {
constexpr int FEAT   = 40960;
constexpr int HALF   = 256;   // per-perspective hidden width
constexpr int CONCAT = 512;
constexpr int BATCH  = 4096;
constexpr int KSLAB  = 64;    // K staged per LDS slab = 2 WMMA k-steps
constexpr int NT     = 16;    // 16 N-tiles of 16 -> all 256 cols of one perspective
constexpr int WAVES  = 4;
constexpr int BLOCK  = WAVES * 32;
constexpr int MB     = WAVES * 16;  // 64 batch rows per block
constexpr int LROW64 = 36;    // LDS row stride in dwords (128B bf16 data + 16B pad)
}

__device__ __forceinline__ unsigned int pack_bf16(float lo, float hi) {
  unsigned short l = __builtin_bit_cast(unsigned short, (__bf16)lo);
  unsigned short h = __builtin_bit_cast(unsigned short, (__bf16)hi);
  return (unsigned int)l | ((unsigned int)h << 16);
}

__device__ __forceinline__ float cclamp(float x) {
  return __builtin_fminf(__builtin_fmaxf(x, 0.0f), 1.0f);
}

// Stage w_in[0:256, kb:kb+64) into one LDS buffer as bf16 (convert once per block).
__device__ __forceinline__ void stage_slab(const float* __restrict__ w_in, int kb,
                                           unsigned int* __restrict__ buf, int tid) {
#pragma unroll
  for (int rr = 0; rr < HALF / BLOCK; ++rr) {          // 2 rows per thread
    const int r = tid + rr * BLOCK;
    const float* src = w_in + (size_t)r * FEAT + kb;
    unsigned int* dst = buf + r * LROW64;
#pragma unroll
    for (int g = 0; g < 8; ++g) {                      // 8 chunks of 8 floats
      f32x4 f0 = *(const f32x4*)(src + 8 * g);
      f32x4 f1 = *(const f32x4*)(src + 8 * g + 4);
      u32x4 o;
      o.x = pack_bf16(f0.x, f0.y);
      o.y = pack_bf16(f0.z, f0.w);
      o.z = pack_bf16(f1.x, f1.y);
      o.w = pack_bf16(f1.z, f1.w);
      *(u32x4*)(dst + 4 * g) = o;
    }
  }
}

// Consume one 64-K slab: 2 WMMA k-steps x 16 N-tiles, B-fragments
// software-pipelined through 2 register buffers (breaks dscnt==0 stalls).
__device__ __forceinline__ void compute_slab(v8f acc[NT],
                                             const float* __restrict__ arow,
                                             int kb,
                                             const unsigned int* __restrict__ buf,
                                             int l16, int hi) {
  // A fragments (16x32 bf16) for both k-steps, per ISA 7.12.2 layout:
  // VGPR0-3: K = hi*8 + 0..7 ; VGPR4-7: K = 16 + hi*8 + 0..7
  v16bf afr[2];
#pragma unroll
  for (int ks = 0; ks < 2; ++ks) {
    const float* ap = arow + kb + ks * 32 + hi * 8;
    f32x4 a0 = *(const f32x4*)(ap);
    f32x4 a1 = *(const f32x4*)(ap + 4);
    f32x4 a2 = *(const f32x4*)(ap + 16);
    f32x4 a3 = *(const f32x4*)(ap + 20);
    v16bf a;
    a[0]  = (__bf16)a0.x; a[1]  = (__bf16)a0.y;
    a[2]  = (__bf16)a0.z; a[3]  = (__bf16)a0.w;
    a[4]  = (__bf16)a1.x; a[5]  = (__bf16)a1.y;
    a[6]  = (__bf16)a1.z; a[7]  = (__bf16)a1.w;
    a[8]  = (__bf16)a2.x; a[9]  = (__bf16)a2.y;
    a[10] = (__bf16)a2.z; a[11] = (__bf16)a2.w;
    a[12] = (__bf16)a3.x; a[13] = (__bf16)a3.y;
    a[14] = (__bf16)a3.z; a[15] = (__bf16)a3.w;
    afr[ks] = a;
  }

  union BFRAG { u32x4 u[2]; v16bf v; };
  // B frag address: row = t*16 + l16 ; dword off = ks*16 (K+32) + hi*4 (K+8), second half +8 (K+16)
  const unsigned int* bbase = buf + l16 * LROW64 + hi * 4;

  BFRAG bf[2];
  bf[0].u[0] = *(const u32x4*)(bbase);
  bf[0].u[1] = *(const u32x4*)(bbase + 8);

#pragma unroll
  for (int i = 0; i < 32; ++i) {                 // i = ks*16 + t
    const int ks = i >> 4;
    const int t  = i & 15;
    if (i + 1 < 32) {                            // prefetch next B fragment
      const int i2 = i + 1;
      const unsigned int* bp =
          bbase + (i2 & 15) * (16 * LROW64) + (i2 >> 4) * 16;
      bf[i2 & 1].u[0] = *(const u32x4*)(bp);
      bf[i2 & 1].u[1] = *(const u32x4*)(bp + 8);
    }
    acc[t] = __builtin_amdgcn_wmma_f32_16x16x32_bf16(
        false, afr[ks], false, bf[i & 1].v, (short)0, acc[t], false, false);
  }
}

// ---------------------------------------------------------------------------
// Input layer: act1[b, persp*256 + n] = clip(feat[b,:] . w_in[n,:] + b_in[n])
// Wave tile: 16 M x 256 N. Features stream from HBM exactly once; w_in slabs
// double-buffered in LDS as bf16, staging overlapped with WMMA compute.
// ---------------------------------------------------------------------------
__global__ __launch_bounds__(BLOCK)
void nnue_input_layer_wmma_bf16(const float* __restrict__ wfeat,
                                const float* __restrict__ bfeat,
                                const float* __restrict__ w_in,
                                const float* __restrict__ b_in,
                                float* __restrict__ act1) {
  __shared__ unsigned int ldsB[2][HALF * LROW64];  // 2 x 36 KB

  const int persp = blockIdx.y;
  const float* __restrict__ feat = persp ? bfeat : wfeat;

  const int tid  = threadIdx.x;
  const int lane = tid & 31;
  const int l16  = lane & 15;
  const int hi   = lane >> 4;          // K-subgroup select per fragment layout
  const int wave = tid >> 5;
  const int mbase = blockIdx.x * MB + wave * 16;

  const float* __restrict__ arow = feat + (size_t)(mbase + l16) * FEAT;

  v8f zero = {};
  v8f acc[NT];
#pragma unroll
  for (int t = 0; t < NT; ++t) acc[t] = zero;

  stage_slab(w_in, 0, &ldsB[0][0], tid);
  __syncthreads();

  for (int kb = 0; kb < FEAT; kb += 2 * KSLAB) {
    // stage next slab into buf1 while computing from buf0
    stage_slab(w_in, kb + KSLAB, &ldsB[1][0], tid);
    compute_slab(acc, arow, kb, &ldsB[0][0], l16, hi);
    __syncthreads();

    // stage slab after next into buf0 while computing from buf1
    if (kb + 2 * KSLAB < FEAT)
      stage_slab(w_in, kb + 2 * KSLAB, &ldsB[0][0], tid);
    compute_slab(acc, arow, kb + KSLAB, &ldsB[1][0], l16, hi);
    __syncthreads();
  }

  // epilogue: bias + clipped relu; C layout: VGPR r -> M = r + 8*hi, lane -> N
  const int colbase = persp * HALF;
#pragma unroll
  for (int t = 0; t < NT; ++t) {
    const int n = t * 16 + l16;
    const float bias = b_in[n];
#pragma unroll
    for (int r = 0; r < 8; ++r) {
      const int m = mbase + r + 8 * hi;
      act1[(size_t)m * CONCAT + colbase + n] = cclamp(acc[t][r] + bias);
    }
  }
}

// ---------------------------------------------------------------------------
// Tail: 512 -> 32 -> 32 -> 1 (negligible FLOPs). One thread per batch row.
// ---------------------------------------------------------------------------
__global__ __launch_bounds__(256)
void nnue_tail(const float* __restrict__ act1,
               const float* __restrict__ w_h1, const float* __restrict__ b_h1,
               const float* __restrict__ w_h2, const float* __restrict__ b_h2,
               const float* __restrict__ w_out, const float* __restrict__ b_out,
               float* __restrict__ out) {
  const int row = blockIdx.x * 256 + threadIdx.x;
  const float* a = act1 + (size_t)row * CONCAT;

  float h1[32];
#pragma unroll
  for (int j = 0; j < 32; ++j) h1[j] = b_h1[j];

  for (int k = 0; k < CONCAT; k += 4) {
    f32x4 av = *(const f32x4*)(a + k);
#pragma unroll
    for (int j = 0; j < 32; ++j) {
      f32x4 wv = *(const f32x4*)(w_h1 + j * CONCAT + k);  // uniform -> scalar loads
      h1[j] = __builtin_fmaf(av.x, wv.x, h1[j]);
      h1[j] = __builtin_fmaf(av.y, wv.y, h1[j]);
      h1[j] = __builtin_fmaf(av.z, wv.z, h1[j]);
      h1[j] = __builtin_fmaf(av.w, wv.w, h1[j]);
    }
  }
#pragma unroll
  for (int j = 0; j < 32; ++j) h1[j] = cclamp(h1[j]);

  float h2[32];
#pragma unroll
  for (int j = 0; j < 32; ++j) {
    float s = b_h2[j];
#pragma unroll
    for (int i = 0; i < 32; ++i) s = __builtin_fmaf(h1[i], w_h2[j * 32 + i], s);
    h2[j] = cclamp(s);
  }

  float o = b_out[0];
#pragma unroll
  for (int i = 0; i < 32; ++i) o = __builtin_fmaf(h2[i], w_out[i], o);
  out[row] = o;
}

// ---------------------------------------------------------------------------
extern "C" void kernel_launch(void* const* d_in, const int* in_sizes, int n_in,
                              void* d_out, int out_size, void* d_ws, size_t ws_size,
                              hipStream_t stream) {
  (void)in_sizes; (void)n_in; (void)out_size; (void)ws_size;
  const float* wfeat = (const float*)d_in[0];
  const float* bfeat = (const float*)d_in[1];
  const float* w_in  = (const float*)d_in[2];
  const float* b_in  = (const float*)d_in[3];
  const float* w_h1  = (const float*)d_in[4];
  const float* b_h1  = (const float*)d_in[5];
  const float* w_h2  = (const float*)d_in[6];
  const float* b_h2  = (const float*)d_in[7];
  const float* w_out = (const float*)d_in[8];
  const float* b_out = (const float*)d_in[9];
  float* out  = (float*)d_out;
  float* act1 = (float*)d_ws;  // [BATCH][512] f32 = 8 MB scratch

  dim3 grid1(BATCH / MB, 2);
  nnue_input_layer_wmma_bf16<<<grid1, BLOCK, 0, stream>>>(wfeat, bfeat, w_in, b_in, act1);
  nnue_tail<<<BATCH / 256, 256, 0, stream>>>(act1, w_h1, b_h1, w_h2, b_h2, w_out, b_out, out);
}
